// UpdateHidden_39857296507340
// MI455X (gfx1250) — matrix operand order repaired and checked by reading, not compiled, not run on hardware
//
#include <hip/hip_runtime.h>
#include <stdint.h>
#include <math.h>

// ---------------------------------------------------------------------------
// Problem constants
// ---------------------------------------------------------------------------
#define E_TOT   200000
#define LDIM    128

typedef __attribute__((ext_vector_type(16))) __bf16 v16bf;
typedef __attribute__((ext_vector_type(8)))  float  v8f;

union FragB { uint32_t u[8]; v16bf v; };

__device__ __forceinline__ uint16_t f2bf(float x) {
  union { float f; uint32_t u; } c; c.f = x;
  uint32_t r = ((c.u >> 16) & 1u) + 0x7fffu;
  return (uint16_t)((c.u + r) >> 16);
}
__device__ __forceinline__ float sigmoidf_(float x) { return 1.f / (1.f + __expf(-x)); }
__device__ __forceinline__ float siluf_(float x)    { return x * sigmoidf_(x); }

// ---------------------------------------------------------------------------
// Weight fragment workspace layout (uint32 words). Each (kt,nt) B-fragment is
// 256 words: word = frag*256 + v*32 + lane, packing bf16 pair (K even, K odd).
// ---------------------------------------------------------------------------
enum {
  OFF_WM0    = 0,        // 192x128: 6*8  *256 = 12288
  OFF_WM1R   = 12288,    // 64x32:   2*2  *256 =  1024
  OFF_WM1I   = 13312,
  OFF_WM1IN  = 14336,    // negated W_m1i
  OFF_WLAT   = 15360,    // 128x160: 4*10 *256 = 10240
  OFF_WSPOST = 25600,    // 64x64:   2*4  *256 =  2048
  OFF_WVPOST = 27648,    // 32x32:   1*2  *256 =   512
  OFF_HIDW   = 28160,    // 128x96:  4*6  *256 =  6144
  OFF_LATW0  = 34304,    // 200x128: 7*8  *256 = 14336 (K padded to 224)
  OFF_LATW1  = 48640,    // 128x128: 4*8  *256 =  8192
  OFF_LATW2  = 56832,
  OFF_WSRES  = 65024,
  OFF_WVRES  = 67072,
  WS_WORDS   = 67584     // ~270 KB, L2-resident
};

__global__ void pack_b_frags(const float* __restrict__ src, uint32_t* __restrict__ dst,
                             int K, int N, int Nt, int total, float scale) {
  int idx = blockIdx.x * blockDim.x + threadIdx.x;
  if (idx >= total) return;
  int lane = idx & 31, v = (idx >> 5) & 7, f = idx >> 8;
  int nt = f % Nt, kt = f / Nt;
  int n  = nt * 16 + (lane & 15);
  int kl = ((lane & 16) ? 8 : 0) + 2 * (v & 3) + ((v & 4) ? 16 : 0);
  int k0 = kt * 32 + kl;
  float a = (k0     < K && n < N) ? src[(size_t)k0 * N + n] * scale : 0.f;
  float b = (k0 + 1 < K && n < N) ? src[(size_t)(k0 + 1) * N + n] * scale : 0.f;
  dst[idx] = (uint32_t)f2bf(a) | ((uint32_t)f2bf(b) << 16);
}

// ---------------------------------------------------------------------------
// WMMA helpers (wave32, 16x16x32 bf16 -> f32)
// ---------------------------------------------------------------------------
__device__ __forceinline__ v16bf load_a_frag(const uint16_t* A, int Kpad, int kt, int lane) {
  int m  = lane & 15;
  int kb = (lane & 16) ? 8 : 0;
  const uint32_t* row = (const uint32_t*)(A + m * Kpad);
  int base = (kt * 32 + kb) >> 1;
  FragB f;
#pragma unroll
  for (int v = 0; v < 4; ++v) f.u[v] = row[base + v];
#pragma unroll
  for (int v = 0; v < 4; ++v) f.u[4 + v] = row[base + 8 + v];
  return f.v;
}

__device__ __forceinline__ v16bf load_b_frag(const uint32_t* frag, int lane) {
  FragB f;
#pragma unroll
  for (int v = 0; v < 8; ++v) f.u[v] = frag[v * 32 + lane];
  return f.v;
}

template <int NT>
__device__ __forceinline__ void wave_gemm(const uint16_t* A, int Kpad,
                                          const uint32_t* B, int KT,
                                          v8f* acc, int lane, bool init) {
  if (init) {
    v8f z = {0.f, 0.f, 0.f, 0.f, 0.f, 0.f, 0.f, 0.f};
#pragma unroll
    for (int i = 0; i < NT; ++i) acc[i] = z;
  }
  for (int kt = 0; kt < KT; ++kt) {
    v16bf a = load_a_frag(A, Kpad, kt, lane);
    const uint32_t* bk = B + (size_t)kt * NT * 256;
#pragma unroll
    for (int nt = 0; nt < NT; ++nt) {
      v16bf b = load_b_frag(bk + nt * 256, lane);
      acc[nt] = __builtin_amdgcn_wmma_f32_16x16x32_bf16(
          false, a, false, b, (short)0, acc[nt], false, false);
    }
  }
}

// ---------------------------------------------------------------------------
// LDS arena (bytes) — lifetime-overlapped regions, 1 wave / 16-edge tile
// ---------------------------------------------------------------------------
#define A_META   0        // ctr/nbr/cut/R/stats/ae
#define A_LA32   1280     // 16x128 f32                       (whole kernel)
#define A_LA16   9472     // 16x128 bf16                      (s GEMM)
#define A_LNLA   13568    // 16x128 bf16 ln(la)               (w GEMM)
#define A_HSTMP  17664    // 16x64  f32 raw hs (early)  -> GATES(2K)+OZ(2K)
#define A_GATES  17664
#define A_OZ     19712
#define A_HV32   21760    // 16x96  f32 raw hv (early)  -> AV16 (3K)
#define A_AV16   21760
#define A_AM0    27904    // 16x192 bf16 m0 A           -> ALIN (7K, 16x224)
#define A_ALIN   27904
#define A_XM     34048    // 16x64  bf16
#define A_YM     36096    // 16x64  bf16
#define A_HS16   38144    // 16x64  bf16 raw hs (residual)
#define A_HV016  40192    // 3x16x32 bf16 raw hv (residual)
#define A_S32    43264    // 16x160 f32 s               -> W32(6K) + H16(4K)
#define A_W32    43264
#define A_H16    49408
#define A_OX     53504    // 16x32 f32                  -> HS2 (4K) later
#define A_OY     55552
#define A_HS2    53504
#define A_SCA16  57600    // 16x64 bf16
#define ARENA_SZ 59648

struct KParams {
  const float *latents, *node_features, *hidden, *onehot, *edge_vec, *cutoff;
  const int   *edge_index, *active_edges;
  const float *sln_ws, *sln_bs, *sln_wv, *lnW, *lnB, *lnoW, *lnoB, *bs_post, *bs_res;
  const uint32_t* wf;
  float* out;
};

__launch_bounds__(32)
__global__ void edge_update_kernel(KParams p) {
  __shared__ __align__(16) char arena[ARENA_SZ];
  const int lane = threadIdx.x;
  const int e0   = blockIdx.x * 16;

  int*   ctr_s = (int*)(arena + A_META + 0);
  int*   nbr_s = (int*)(arena + A_META + 64);
  float* cut_s = (float*)(arena + A_META + 128);
  float* Rm    = (float*)(arena + A_META + 192);   // 16x9
  float* mu_hs = (float*)(arena + A_META + 768);
  float* rs_hs = (float*)(arena + A_META + 832);
  float* rs_hv = (float*)(arena + A_META + 896);
  float* mu_la = (float*)(arena + A_META + 960);
  float* rs_la = (float*)(arena + A_META + 1024);
  int*   ae_s  = (int*)(arena + A_META + 1088);

  float*    LA32 = (float*)(arena + A_LA32);
  uint16_t* LA16 = (uint16_t*)(arena + A_LA16);
  uint16_t* LNLA = (uint16_t*)(arena + A_LNLA);
  float*    HSTMP= (float*)(arena + A_HSTMP);
  float*    GATES= (float*)(arena + A_GATES);
  float*    OZ   = (float*)(arena + A_OZ);
  float*    HV32 = (float*)(arena + A_HV32);
  uint16_t* AV16 = (uint16_t*)(arena + A_AV16);
  uint16_t* AM0  = (uint16_t*)(arena + A_AM0);
  uint16_t* ALIN = (uint16_t*)(arena + A_ALIN);
  uint16_t* XM16 = (uint16_t*)(arena + A_XM);
  uint16_t* YM16 = (uint16_t*)(arena + A_YM);
  uint16_t* HS16 = (uint16_t*)(arena + A_HS16);
  uint16_t* HV016= (uint16_t*)(arena + A_HV016);
  float*    S32  = (float*)(arena + A_S32);
  float*    W32  = (float*)(arena + A_W32);
  uint16_t* H16  = (uint16_t*)(arena + A_H16);
  float*    OX   = (float*)(arena + A_OX);
  float*    OY   = (float*)(arena + A_OY);
  float*    HS2  = (float*)(arena + A_HS2);
  uint16_t* SCA16= (uint16_t*)(arena + A_SCA16);

  const int mb = (lane & 16) ? 8 : 0;
  const int nl = lane & 15;

  // ---------------- per-edge meta + rotation frame ----------------
  if (lane < 16) {
    int eg = e0 + lane;
    int ae = p.active_edges[eg];
    ae_s[lane]  = ae;
    ctr_s[lane] = p.edge_index[ae];
    nbr_s[lane] = p.edge_index[E_TOT + ae];
    cut_s[lane] = p.cutoff[ae];
    float vx = p.edge_vec[3 * (size_t)ae + 0];
    float vy = p.edge_vec[3 * (size_t)ae + 1];
    float vz = p.edge_vec[3 * (size_t)ae + 2];
    float inv = rsqrtf(vx * vx + vy * vy + vz * vz);
    float nx = vx * inv, ny = vy * inv, nz = vz * inv;
    float ax, ay;
    if (fabsf(nx) < 0.9f) { ax = 1.f; ay = 0.f; } else { ax = 0.f; ay = 1.f; }
    float r1x = ay * nz, r1y = -ax * nz, r1z = ax * ny - ay * nx;
    inv = rsqrtf(r1x * r1x + r1y * r1y + r1z * r1z);
    r1x *= inv; r1y *= inv; r1z *= inv;
    float r2x = ny * r1z - nz * r1y;
    float r2y = nz * r1x - nx * r1z;
    float r2z = nx * r1y - ny * r1x;
    float* R = Rm + lane * 9;
    R[0] = r1x; R[1] = r1y; R[2] = r1z;
    R[3] = r2x; R[4] = r2y; R[5] = r2z;
    R[6] = nx;  R[7] = ny;  R[8] = nz;
  }
  __syncthreads();

  // ---------------- gathers ----------------
  for (int i = lane; i < 16 * 128; i += 32) {
    int e = i >> 7, c = i & 127;
    float v = p.latents[(size_t)ae_s[e] * LDIM + c];
    LA32[i] = v; LA16[i] = f2bf(v);
  }
  for (int i = lane; i < 16 * 64; i += 32) {
    int e = i >> 6;
    float v = p.hidden[(size_t)(e0 + e) * 160 + (i & 63)];
    HSTMP[i] = v; HS16[i] = f2bf(v);
  }
  for (int i = lane; i < 16 * 96; i += 32) {
    int e = i / 96, r = i % 96, ch = r / 3, cp = r % 3;
    float v = p.hidden[(size_t)(e0 + e) * 160 + 64 + r];
    HV32[i] = v;
    HV016[cp * 512 + e * 32 + ch] = f2bf(v);
  }
  __syncthreads();

  // ---------------- norms ----------------
  if (lane < 16) {
    float mu = 0.f;
    for (int c = 0; c < 64; ++c) mu += HSTMP[lane * 64 + c];
    mu *= (1.f / 64.f);
    float var = 0.f;
    for (int c = 0; c < 64; ++c) { float d = HSTMP[lane * 64 + c] - mu; var += d * d; }
    mu_hs[lane] = mu; rs_hs[lane] = rsqrtf(var * (1.f / 64.f) + 1e-5f);
    float vv = 0.f;
    for (int c = 0; c < 96; ++c) { float d = HV32[lane * 96 + c]; vv += d * d; }
    rs_hv[lane] = rsqrtf(vv * (1.f / 96.f) + 1e-5f);
    float ml = 0.f;
    for (int c = 0; c < 128; ++c) ml += LA32[lane * 128 + c];
    ml *= (1.f / 128.f);
    float vl = 0.f;
    for (int c = 0; c < 128; ++c) { float d = LA32[lane * 128 + c] - ml; vl += d * d; }
    mu_la[lane] = ml; rs_la[lane] = rsqrtf(vl * (1.f / 128.f) + 1e-5f);
  }
  __syncthreads();

  // A_m0[:,0:64] = nf scalar(ctr); [:,64:128] = LN(hs); LNLA = ln(la)*lnW+lnB
  for (int i = lane; i < 16 * 64; i += 32) {
    int e = i >> 6, c = i & 63;
    AM0[e * 192 + c] = f2bf(p.node_features[(size_t)ctr_s[e] * 160 + c]);
    float hn = (HSTMP[i] - mu_hs[e]) * rs_hs[e] * p.sln_ws[c] + p.sln_bs[c];
    AM0[e * 192 + 64 + c] = f2bf(hn);
  }
  for (int i = lane; i < 16 * 128; i += 32) {
    int e = i >> 7, c = i & 127;
    LNLA[i] = f2bf((LA32[i] - mu_la[e]) * rs_la[e] * p.lnW[c] + p.lnB[c]);
  }
  __syncthreads();

  // rotate xv=[nf_vec | hv_n] -> XM, YM, AM0[:,128:192]
  for (int i = lane; i < 16 * 64; i += 32) {
    int e = i >> 6, ch = i & 63;
    float x, y, z;
    if (ch < 32) {
      const float* nv = p.node_features + (size_t)ctr_s[e] * 160 + 64 + ch * 3;
      x = nv[0]; y = nv[1]; z = nv[2];
    } else {
      int c2 = ch - 32;
      float rs = rs_hv[e] * p.sln_wv[c2];
      const float* hp = HV32 + e * 96 + c2 * 3;
      x = hp[0] * rs; y = hp[1] * rs; z = hp[2] * rs;
    }
    const float* R = Rm + e * 9;
    XM16[e * 64 + ch] = f2bf(R[0] * x + R[1] * y + R[2] * z);
    YM16[e * 64 + ch] = f2bf(R[3] * x + R[4] * y + R[5] * z);
    AM0[e * 192 + 128 + ch] = f2bf(R[6] * x + R[7] * y + R[8] * z);
  }
  __syncthreads();

  // ---------------- s = la @ W_lat/sqrt(128)  [16x160] ----------------
  {
    v8f acc[10];
    wave_gemm<10>(LA16, 128, p.wf + OFF_WLAT, 4, acc, lane, true);
#pragma unroll
    for (int nt = 0; nt < 10; ++nt)
#pragma unroll
      for (int r = 0; r < 8; ++r)
        S32[(mb + r) * 160 + nt * 16 + nl] = acc[nt][r];
  }
  __syncthreads();

  // ---------------- complex m1: ox, oy (scaled by mv) ----------------
  {
    v8f acc[2];
    wave_gemm<2>(XM16, 64, p.wf + OFF_WM1R,  2, acc, lane, true);
    wave_gemm<2>(YM16, 64, p.wf + OFF_WM1IN, 2, acc, lane, false);
#pragma unroll
    for (int nt = 0; nt < 2; ++nt)
#pragma unroll
      for (int r = 0; r < 8; ++r) {
        int m = mb + r, n = nt * 16 + nl;
        OX[m * 32 + n] = acc[nt][r] * S32[m * 160 + 128 + n];
      }
    wave_gemm<2>(YM16, 64, p.wf + OFF_WM1R, 2, acc, lane, true);
    wave_gemm<2>(XM16, 64, p.wf + OFF_WM1I, 2, acc, lane, false);
#pragma unroll
    for (int nt = 0; nt < 2; ++nt)
#pragma unroll
      for (int r = 0; r < 8; ++r) {
        int m = mb + r, n = nt * 16 + nl;
        OY[m * 32 + n] = acc[nt][r] * S32[m * 160 + 128 + n];
      }
  }
  __syncthreads();

  // ---------------- out0 = m0_in @ W_m0/sqrt(192), scaled by s ----------------
  {
    v8f acc[8];
    wave_gemm<8>(AM0, 192, p.wf + OFF_WM0, 6, acc, lane, true);
#pragma unroll
    for (int nt = 0; nt < 8; ++nt)
#pragma unroll
      for (int r = 0; r < 8; ++r) {
        int m = mb + r, n = nt * 16 + nl;
        float v = acc[nt][r] * S32[m * 160 + n];
        if (nt < 4)      SCA16[m * 64 + n]       = f2bf(siluf_(v));
        else if (nt < 6) GATES[m * 32 + (n - 64)] = sigmoidf_(v);
        else             OZ[m * 32 + (n - 96)]    = v;
      }
  }
  __syncthreads();

  // ---------------- v = R^T @ (ox,oy,oz) * gates ----------------
  for (int i = lane; i < 16 * 32; i += 32) {
    int e = i >> 5;
    float ox = OX[i], oy = OY[i], oz = OZ[i], g = GATES[i];
    const float* R = Rm + e * 9;
    AV16[0 * 512 + i] = f2bf((R[0] * ox + R[3] * oy + R[6] * oz) * g);
    AV16[1 * 512 + i] = f2bf((R[1] * ox + R[4] * oy + R[7] * oz) * g);
    AV16[2 * 512 + i] = f2bf((R[2] * ox + R[5] * oy + R[8] * oz) * g);
  }
  __syncthreads();

  // ---------------- hs2 = silu(out0_s) @ Ws_post/sqrt(64) + bs_post ----------------
  {
    v8f acc[4];
    wave_gemm<4>(SCA16, 64, p.wf + OFF_WSPOST, 2, acc, lane, true);
#pragma unroll
    for (int nt = 0; nt < 4; ++nt)
#pragma unroll
      for (int r = 0; r < 8; ++r) {
        int m = mb + r, n = nt * 16 + nl;
        HS2[m * 64 + n] = acc[nt][r] + p.bs_post[n];
      }
  }
  __syncthreads();

  // ---------------- w = ln(la) @ hid_W/sqrt(128)  [16x96] ----------------
  {
    v8f acc[6];
    wave_gemm<6>(LNLA, 128, p.wf + OFF_HIDW, 4, acc, lane, true);
#pragma unroll
    for (int nt = 0; nt < 6; ++nt)
#pragma unroll
      for (int r = 0; r < 8; ++r)
        W32[(mb + r) * 96 + nt * 16 + nl] = acc[nt][r];
  }
  __syncthreads();

  // ---------------- hidden_out ----------------
  const float c_old = 0.894427190999916f;   // 1/sqrt(1.25)
  const float c_new = 0.447213595499958f;   // 0.5*c_old
  {   // scalar part
    v8f accr[4];
    wave_gemm<4>(HS16, 64, p.wf + OFF_WSRES, 2, accr, lane, true);
#pragma unroll
    for (int nt = 0; nt < 4; ++nt)
#pragma unroll
      for (int r = 0; r < 8; ++r) {
        int m = mb + r, n = nt * 16 + nl;
        float res = accr[nt][r] + p.bs_res[n];
        p.out[(size_t)(e0 + m) * 160 + n] =
            c_new * HS2[m * 64 + n] * W32[m * 96 + n] + c_old * res;
      }
  }
  for (int cp = 0; cp < 3; ++cp) {   // vector part per component
    v8f a1[2], a2[2];
    wave_gemm<2>(AV16  + cp * 512, 32, p.wf + OFF_WVPOST, 1, a1, lane, true);
    wave_gemm<2>(HV016 + cp * 512, 32, p.wf + OFF_WVRES,  1, a2, lane, true);
#pragma unroll
    for (int nt = 0; nt < 2; ++nt)
#pragma unroll
      for (int r = 0; r < 8; ++r) {
        int m = mb + r, n = nt * 16 + nl;
        float val = c_new * a1[nt][r] * W32[m * 96 + 64 + n] + c_old * a2[nt][r];
        p.out[(size_t)(e0 + m) * 160 + 64 + n * 3 + cp] = val;
      }
  }
  __syncthreads();

  // ---------------- latent MLP ----------------
  for (int i = lane; i < 16 * 224; i += 32) {
    int e = i / 224, c = i % 224;
    float v;
    if (c < 4)        v = p.onehot[(size_t)ctr_s[e] * 4 + c];
    else if (c < 132) { int cc = c - 4;
      v = (LA32[e * 128 + cc] - mu_la[e]) * rs_la[e] * p.lnoW[cc] + p.lnoB[cc]; }
    else if (c < 196) v = HS2[e * 64 + (c - 132)];
    else if (c < 200) v = p.onehot[(size_t)nbr_s[e] * 4 + (c - 196)];
    else              v = 0.f;
    ALIN[i] = f2bf(v);
  }
  __syncthreads();
  {
    v8f acc[8];
    wave_gemm<8>(ALIN, 224, p.wf + OFF_LATW0, 7, acc, lane, true);
#pragma unroll
    for (int nt = 0; nt < 8; ++nt)
#pragma unroll
      for (int r = 0; r < 8; ++r)
        H16[(mb + r) * 128 + nt * 16 + nl] = f2bf(siluf_(acc[nt][r]));
  }
  __syncthreads();
  {
    v8f acc[8];
    wave_gemm<8>(H16, 128, p.wf + OFF_LATW1, 4, acc, lane, true);
    __syncthreads();
#pragma unroll
    for (int nt = 0; nt < 8; ++nt)
#pragma unroll
      for (int r = 0; r < 8; ++r)
        H16[(mb + r) * 128 + nt * 16 + nl] = f2bf(siluf_(acc[nt][r]));
  }
  __syncthreads();
  {
    v8f acc[8];
    wave_gemm<8>(H16, 128, p.wf + OFF_LATW2, 4, acc, lane, true);
#pragma unroll
    for (int nt = 0; nt < 8; ++nt)
#pragma unroll
      for (int r = 0; r < 8; ++r) {
        int m = mb + r, n = nt * 16 + nl;
        float newlat = cut_s[m] * acc[nt][r];
        p.out[(size_t)E_TOT * 160 + (size_t)(e0 + m) * 128 + n] =
            c_new * newlat + c_old * LA32[m * 128 + n];
      }
  }
}

// ---------------------------------------------------------------------------
// Host entry
// ---------------------------------------------------------------------------
extern "C" void kernel_launch(void* const* d_in, const int* in_sizes, int n_in,
                              void* d_out, int out_size, void* d_ws, size_t ws_size,
                              hipStream_t stream) {
  (void)in_sizes; (void)n_in; (void)out_size; (void)ws_size;
  KParams p;
  p.latents       = (const float*)d_in[0];
  p.node_features = (const float*)d_in[1];
  p.hidden        = (const float*)d_in[2];
  p.onehot        = (const float*)d_in[3];
  p.edge_index    = (const int*)d_in[4];
  p.edge_vec      = (const float*)d_in[5];
  p.cutoff        = (const float*)d_in[6];
  p.active_edges  = (const int*)d_in[7];
  p.sln_ws        = (const float*)d_in[8];
  p.sln_bs        = (const float*)d_in[9];
  p.sln_wv        = (const float*)d_in[10];
  const float* W_m0    = (const float*)d_in[11];
  const float* W_m1r   = (const float*)d_in[12];
  const float* W_m1i   = (const float*)d_in[13];
  const float* W_lat   = (const float*)d_in[14];
  p.lnW  = (const float*)d_in[15];
  p.lnB  = (const float*)d_in[16];
  p.lnoW = (const float*)d_in[17];
  p.lnoB = (const float*)d_in[18];
  const float* Ws_post = (const float*)d_in[19];
  p.bs_post = (const float*)d_in[20];
  const float* Wv_post = (const float*)d_in[21];
  const float* hid_W   = (const float*)d_in[22];
  const float* lat_W0  = (const float*)d_in[23];
  const float* lat_W1  = (const float*)d_in[24];
  const float* lat_W2  = (const float*)d_in[25];
  const float* Ws_res  = (const float*)d_in[26];
  p.bs_res = (const float*)d_in[27];
  const float* Wv_res  = (const float*)d_in[28];

  uint32_t* wf = (uint32_t*)d_ws;
  p.wf  = wf;
  p.out = (float*)d_out;

  auto pack = [&](const float* src, int off, int K, int N, int Kt, int Nt, float scale) {
    int total = Kt * Nt * 256;
    pack_b_frags<<<dim3((total + 255) / 256), dim3(256), 0, stream>>>(
        src, wf + off, K, N, Nt, total, scale);
  };
  const float SILU_C = 1.679f;
  pack(W_m0,    OFF_WM0,    192, 128, 6,  8,  1.f / sqrtf(192.f));
  pack(W_m1r,   OFF_WM1R,   64,  32,  2,  2,  1.f / sqrtf(64.f));
  pack(W_m1i,   OFF_WM1I,   64,  32,  2,  2,  1.f / sqrtf(64.f));
  pack(W_m1i,   OFF_WM1IN,  64,  32,  2,  2, -1.f / sqrtf(64.f));
  pack(W_lat,   OFF_WLAT,   128, 160, 4, 10,  1.f / sqrtf(128.f));
  pack(Ws_post, OFF_WSPOST, 64,  64,  2,  4,  1.f / sqrtf(64.f));
  pack(Wv_post, OFF_WVPOST, 32,  32,  1,  2,  1.f / sqrtf(32.f));
  pack(hid_W,   OFF_HIDW,   128, 96,  4,  6,  1.f / sqrtf(128.f));
  pack(lat_W0,  OFF_LATW0,  200, 128, 7,  8,  1.f / sqrtf(200.f));
  pack(lat_W1,  OFF_LATW1,  128, 128, 4,  8,  SILU_C / sqrtf(128.f));
  pack(lat_W2,  OFF_LATW2,  128, 128, 4,  8,  SILU_C / sqrtf(128.f));
  pack(Ws_res,  OFF_WSRES,  64,  64,  2,  4,  1.f / sqrtf(64.f));
  pack(Wv_res,  OFF_WVRES,  32,  32,  1,  2,  1.f / sqrtf(32.f));

  edge_update_kernel<<<dim3(E_TOT / 16), dim3(32), 0, stream>>>(p);
}
